// Attention_75058848465066
// MI455X (gfx1250) — matrix operand order repaired and checked by reading, not compiled, not run on hardware
//
#include <hip/hip_runtime.h>
#include <hip/hip_bf16.h>

typedef __bf16 bf16_t;
typedef __attribute__((ext_vector_type(16))) __bf16 v16bf;
typedef __attribute__((ext_vector_type(8)))  float  v8f;

#define WMMA_BF16(A, B, C) \
  __builtin_amdgcn_wmma_f32_16x16x32_bf16(false, (A), false, (B), (short)0, (C), false, false)

union FragBF { v16bf v; uint4 q[2]; unsigned short us[16]; };

__device__ __forceinline__ bf16_t f2bf(float f) {
  unsigned int u = __builtin_bit_cast(unsigned int, f);
  u += 0x7FFFu + ((u >> 16) & 1u);            // round-to-nearest-even
  unsigned short h = (unsigned short)(u >> 16);
  return __builtin_bit_cast(bf16_t, h);
}

// A-fragment 16x32 bf16: lane (0-15) row m=lane&15; element i <-> K = 16*half + i
__device__ __forceinline__ v16bf load_a_frag(const bf16_t* p, int ld, int lane) {
  const int m = lane & 15, kh = (lane >> 4) << 4;
  FragBF f;
  f.q[0] = *(const uint4*)(p + m * ld + kh);
  f.q[1] = *(const uint4*)(p + m * ld + kh + 8);
  return f.v;
}

// B-fragment 32x16 bf16 from a source stored [N x K] row-major (i.e. B^T): element i <-> K = kh + i
__device__ __forceinline__ v16bf load_bt_frag(const bf16_t* p, int ld, int lane) {
  const int n = lane & 15, kh = (lane >> 4) << 4;
  FragBF f;
  f.q[0] = *(const uint4*)(p + n * ld + kh);
  f.q[1] = *(const uint4*)(p + n * ld + kh + 8);
  return f.v;
}

// B-fragment 32x16 bf16 from a source stored [K x N] row-major
__device__ __forceinline__ v16bf load_b_rm_frag(const bf16_t* p, int ld, int lane) {
  const int n = lane & 15, kh = (lane >> 4) << 4;
  FragBF f;
#pragma unroll
  for (int i = 0; i < 16; ++i)
    f.us[i] = *(const unsigned short*)(p + (kh + i) * ld + n);
  return f.v;
}

// ---------------------------------------------------------------------------
// Prep kernel: bf16-transpose the projection weights so B fragments load as
// contiguous-K b128s.  wqkv_t: [1536 x 512], wout_t: [512 x 512].
// ---------------------------------------------------------------------------
__global__ void convert_weights(const float* __restrict__ wqkv, const float* __restrict__ wout,
                                bf16_t* __restrict__ wqkv_t, bf16_t* __restrict__ wout_t) {
  const int i = blockIdx.x * blockDim.x + threadIdx.x;
  const int n1 = 1536 * 512;
  if (i < n1) {
    const int n = i >> 9, k = i & 511;
    wqkv_t[i] = f2bf(wqkv[k * 1536 + n]);
  } else {
    const int j = i - n1;
    if (j < 512 * 512) {
      const int n = j >> 9, k = j & 511;
      wout_t[j] = f2bf(wout[k * 512 + n]);
    }
  }
}

// ---------------------------------------------------------------------------
// DPB MLP: 289 relative offsets -> scalar bias, then expand to the shared
// 64x64 bias matrix with the reference's stride-15 gather.
// ---------------------------------------------------------------------------
__device__ __forceinline__ void lnrelu128(const float* z, const float* g, float* out) {
  float m = 0.f;
#pragma unroll 4
  for (int j = 0; j < 128; ++j) m += z[j];
  m *= (1.f / 128.f);
  float v = 0.f;
#pragma unroll 4
  for (int j = 0; j < 128; ++j) { float d = z[j] - m; v += d * d; }
  v *= (1.f / 128.f);
  const float r = rsqrtf(v + 1e-5f);
  for (int j = 0; j < 128; ++j) {
    float y = (z[j] - m) * r * g[j];
    out[j] = y > 0.f ? y : 0.f;
  }
}

__global__ void dpb_kernel(const float* __restrict__ w0, const float* __restrict__ b0, const float* __restrict__ ln0,
                           const float* __restrict__ w1, const float* __restrict__ b1, const float* __restrict__ ln1,
                           const float* __restrict__ w2, const float* __restrict__ b2, const float* __restrict__ ln2,
                           const float* __restrict__ w3, const float* __restrict__ b3,
                           float* __restrict__ bias_mat) {
  __shared__ float biases[289];
  const int t = threadIdx.x;
  if (t < 289) {
    const float p0 = (float)(t / 17 - 8), p1 = (float)(t % 17 - 8);
    float z[128], h[128];
    for (int j = 0; j < 128; ++j) z[j] = p0 * w0[j] + p1 * w0[128 + j] + b0[j];
    lnrelu128(z, ln0, h);
    for (int j = 0; j < 128; ++j) {
      float a = b1[j];
      for (int i = 0; i < 128; ++i) a += h[i] * w1[i * 128 + j];
      z[j] = a;
    }
    lnrelu128(z, ln1, h);
    for (int j = 0; j < 128; ++j) {
      float a = b2[j];
      for (int i = 0; i < 128; ++i) a += h[i] * w2[i * 128 + j];
      z[j] = a;
    }
    lnrelu128(z, ln2, h);
    float bias = b3[0];
    for (int j = 0; j < 128; ++j) bias += h[j] * w3[j];
    biases[t] = bias;
  }
  __syncthreads();
  for (int i = t; i < 4096; i += blockDim.x) {
    const int ti = i >> 6, tj = i & 63;
    const int y1 = ti >> 3, x1 = ti & 7, y2 = tj >> 3, x2 = tj & 7;
    const int idx = (y1 - y2 + 7) * 15 + (x1 - x2 + 7);   // reference's exact gather
    bias_mat[i] = biases[idx];
  }
}

// ---------------------------------------------------------------------------
// Fused per-window kernel: one workgroup (256 threads = 8 waves) per 8x8
// window.  LDS: [0,64K) xln bf16 (reused as attn scratch), [64K,256K) qkv
// bf16 (q columns reused for attention output), [256K,272K) bias f32.
// ---------------------------------------------------------------------------
__global__ __launch_bounds__(256)
void fused_window_attn(const float* __restrict__ x,
                       const float* __restrict__ norm_g, const float* __restrict__ norm_b,
                       const bf16_t* __restrict__ wqkv_t, const bf16_t* __restrict__ wout_t,
                       const float* __restrict__ bias_mat, const float* __restrict__ b_out,
                       float* __restrict__ out) {
  extern __shared__ char smem[];
  bf16_t* xln  = (bf16_t*)smem;                       // 64 x 512 bf16 (64 KB)
  bf16_t* qkv  = (bf16_t*)(smem + 65536);             // 64 x 1536 bf16 (192 KB)
  float*  biasl = (float*)(smem + 65536 + 196608);    // 64 x 64 f32  (16 KB)

  const int tid  = threadIdx.x;
  const int lane = tid & 31;
  const int wv   = tid >> 5;
  const int n    = lane & 15;
  const int lh   = lane >> 4;
  const int blk  = blockIdx.x;                        // window id 0..1023
  const int b    = blk >> 6, bh = (blk >> 3) & 7, bw = blk & 7;
  const float scale = 0.17677669529663687f;           // 32^-0.5

  // ---- phase 0: stage bias matrix -----------------------------------------
  for (int i = tid; i < 4096; i += 256) biasl[i] = bias_mat[i];

  // ---- phase 1: LayerNorm -> bf16 into LDS (4 threads per token) ----------
  {
    const int tok = tid >> 2, part = tid & 3;
    const int s1 = tok >> 3, s2 = tok & 7;
    const float* xrow = x + (size_t)(((b * 64 + bh * 8 + s1) * 64) + bw * 8 + s2) * 512;
    const float4* xp = (const float4*)xrow + part * 32;
    float4 vb[32];
    float s = 0.f, ss = 0.f;
#pragma unroll 8
    for (int i = 0; i < 32; ++i) {
      float4 t4 = xp[i];
      vb[i] = t4;
      s  += t4.x + t4.y + t4.z + t4.w;
      ss += t4.x * t4.x + t4.y * t4.y + t4.z * t4.z + t4.w * t4.w;
    }
    s  += __shfl_xor(s, 1, 32);  s  += __shfl_xor(s, 2, 32);
    ss += __shfl_xor(ss, 1, 32); ss += __shfl_xor(ss, 2, 32);
    const float mean = s * (1.f / 512.f);
    const float var  = ss * (1.f / 512.f) - mean * mean;
    const float rstd = rsqrtf(var + 1e-5f);
#pragma unroll 8
    for (int i = 0; i < 32; ++i) {
      const int c = part * 128 + i * 4;
      float4 t4 = vb[i];
      xln[tok * 512 + c + 0] = f2bf((t4.x - mean) * rstd * norm_g[c + 0] + norm_b[c + 0]);
      xln[tok * 512 + c + 1] = f2bf((t4.y - mean) * rstd * norm_g[c + 1] + norm_b[c + 1]);
      xln[tok * 512 + c + 2] = f2bf((t4.z - mean) * rstd * norm_g[c + 2] + norm_b[c + 2]);
      xln[tok * 512 + c + 3] = f2bf((t4.w - mean) * rstd * norm_g[c + 3] + norm_b[c + 3]);
    }
  }
  __syncthreads();

  // ---- phase 2: QKV GEMM  qkv[64,1536] = xln[64,512] @ Wqkv ---------------
  {
    const int nbase = wv * 192;                       // wave owns 192 of 1536 cols
    for (int nc = 0; nc < 3; ++nc) {
      v8f acc[4][4];
#pragma unroll
      for (int m = 0; m < 4; ++m)
#pragma unroll
        for (int nt = 0; nt < 4; ++nt) acc[m][nt] = (v8f)0.f;
      for (int k0 = 0; k0 < 512; k0 += 32) {
        v16bf afr[4];
#pragma unroll
        for (int m = 0; m < 4; ++m) afr[m] = load_a_frag(xln + m * 16 * 512 + k0, 512, lane);
#pragma unroll
        for (int nt = 0; nt < 4; ++nt) {
          const int n0 = nbase + nc * 64 + nt * 16;
          v16bf bfr = load_bt_frag(wqkv_t + (size_t)n0 * 512 + k0, 512, lane);
#pragma unroll
          for (int m = 0; m < 4; ++m) acc[m][nt] = WMMA_BF16(afr[m], bfr, acc[m][nt]);
        }
      }
#pragma unroll
      for (int nt = 0; nt < 4; ++nt) {
        const int n0 = nbase + nc * 64 + nt * 16;
#pragma unroll
        for (int m = 0; m < 4; ++m)
#pragma unroll
          for (int r = 0; r < 8; ++r)
            qkv[(m * 16 + r + 8 * lh) * 1536 + n0 + n] = f2bf(acc[m][nt][r]);
      }
    }
  }
  __syncthreads();

  // ---- phase 3: attention, 2 heads per wave --------------------------------
  bf16_t* ascr = ((bf16_t*)smem) + wv * 4096;          // per-wave 64x64 bf16 scratch
  for (int hh = 0; hh < 2; ++hh) {
    const int hd = wv + hh * 8;
    const bf16_t* qp = qkv + hd * 32;
    const bf16_t* kp = qkv + 512 + hd * 32;
    const bf16_t* vp = qkv + 1024 + hd * 32;

    // sim = q @ k^T   (K = 32 = one wmma per 16x16 tile)
    v16bf kb[4];
#pragma unroll
    for (int tj = 0; tj < 4; ++tj) kb[tj] = load_bt_frag(kp + tj * 16 * 1536, 1536, lane);
    v8f sim[4][4];
#pragma unroll
    for (int ti = 0; ti < 4; ++ti) {
      v16bf aq = load_a_frag(qp + ti * 16 * 1536, 1536, lane);
#pragma unroll
      for (int tj = 0; tj < 4; ++tj) {
        sim[ti][tj] = (v8f)0.f;
        sim[ti][tj] = WMMA_BF16(aq, kb[tj], sim[ti][tj]);
      }
    }

    // scale + bias + row softmax (rows live in 16-lane half-waves per C layout)
#pragma unroll
    for (int ti = 0; ti < 4; ++ti) {
#pragma unroll
      for (int r = 0; r < 8; ++r) {
        const int mrow = ti * 16 + r + 8 * lh;
        float mx = -1e30f;
#pragma unroll
        for (int tj = 0; tj < 4; ++tj) {
          float sv = sim[ti][tj][r] * scale + biasl[mrow * 64 + tj * 16 + n];
          sim[ti][tj][r] = sv;
          mx = fmaxf(mx, sv);
        }
        mx = fmaxf(mx, __shfl_xor(mx, 1, 32));
        mx = fmaxf(mx, __shfl_xor(mx, 2, 32));
        mx = fmaxf(mx, __shfl_xor(mx, 4, 32));
        mx = fmaxf(mx, __shfl_xor(mx, 8, 32));
        float sum = 0.f;
#pragma unroll
        for (int tj = 0; tj < 4; ++tj) {
          float e = __expf(sim[ti][tj][r] - mx);
          sim[ti][tj][r] = e;
          sum += e;
        }
        sum += __shfl_xor(sum, 1, 32);
        sum += __shfl_xor(sum, 2, 32);
        sum += __shfl_xor(sum, 4, 32);
        sum += __shfl_xor(sum, 8, 32);
        const float inv = 1.f / sum;
#pragma unroll
        for (int tj = 0; tj < 4; ++tj)
          ascr[mrow * 64 + tj * 16 + n] = f2bf(sim[ti][tj][r] * inv);
      }
    }
    // make this wave's cross-lane LDS writes visible to its own fragment loads
    asm volatile("s_wait_dscnt 0x0" ::: "memory");

    // o = attn @ v   (K = 64 -> two wmma per tile)
    v8f oa[4][2];
#pragma unroll
    for (int ti = 0; ti < 4; ++ti)
#pragma unroll
      for (int tn = 0; tn < 2; ++tn) oa[ti][tn] = (v8f)0.f;
    for (int k0 = 0; k0 < 64; k0 += 32) {
      v16bf vb2[2];
#pragma unroll
      for (int tn = 0; tn < 2; ++tn)
        vb2[tn] = load_b_rm_frag(vp + k0 * 1536 + tn * 16, 1536, lane);
#pragma unroll
      for (int ti = 0; ti < 4; ++ti) {
        v16bf at = load_a_frag(ascr + ti * 16 * 64 + k0, 64, lane);
#pragma unroll
        for (int tn = 0; tn < 2; ++tn) oa[ti][tn] = WMMA_BF16(at, vb2[tn], oa[ti][tn]);
      }
    }
    // write head output over this head's (now dead) q columns
#pragma unroll
    for (int ti = 0; ti < 4; ++ti)
#pragma unroll
      for (int tn = 0; tn < 2; ++tn)
#pragma unroll
        for (int r = 0; r < 8; ++r)
          qkv[(ti * 16 + r + 8 * lh) * 1536 + hd * 32 + tn * 16 + n] = f2bf(oa[ti][tn][r]);
  }
  __syncthreads();

  // ---- phase 4: output projection + bias + de-window store -----------------
  {
    const int nslab = wv * 64;                         // wave owns 64 of 512 cols
    v8f acc[4][4];
#pragma unroll
    for (int m = 0; m < 4; ++m)
#pragma unroll
      for (int tn = 0; tn < 4; ++tn) acc[m][tn] = (v8f)0.f;
    for (int k0 = 0; k0 < 512; k0 += 32) {
      v16bf afr[4];
#pragma unroll
      for (int m = 0; m < 4; ++m) afr[m] = load_a_frag(qkv + m * 16 * 1536 + k0, 1536, lane);
#pragma unroll
      for (int tn = 0; tn < 4; ++tn) {
        v16bf bfr = load_bt_frag(wout_t + (size_t)(nslab + tn * 16) * 512 + k0, 512, lane);
#pragma unroll
        for (int m = 0; m < 4; ++m) acc[m][tn] = WMMA_BF16(afr[m], bfr, acc[m][tn]);
      }
    }
#pragma unroll
    for (int tn = 0; tn < 4; ++tn) {
      const int c = nslab + tn * 16 + n;
      const float bo = b_out[c];
#pragma unroll
      for (int m = 0; m < 4; ++m)
#pragma unroll
        for (int r = 0; r < 8; ++r) {
          const int tok = m * 16 + r + 8 * lh;
          const int s1 = tok >> 3, s2 = tok & 7;
          out[(size_t)(((b * 64 + bh * 8 + s1) * 64) + bw * 8 + s2) * 512 + c] = acc[m][tn][r] + bo;
        }
    }
  }
}

// ---------------------------------------------------------------------------
extern "C" void kernel_launch(void* const* d_in, const int* in_sizes, int n_in,
                              void* d_out, int out_size, void* d_ws, size_t ws_size,
                              hipStream_t stream) {
  (void)in_sizes; (void)n_in; (void)out_size; (void)ws_size;
  const float* x      = (const float*)d_in[0];
  const float* norm_g = (const float*)d_in[1];
  const float* norm_b = (const float*)d_in[2];
  const float* w_qkv  = (const float*)d_in[3];
  const float* w_out  = (const float*)d_in[4];
  const float* b_out  = (const float*)d_in[5];
  const float* dpb_w0 = (const float*)d_in[6];
  const float* dpb_b0 = (const float*)d_in[7];
  const float* dpb_l0 = (const float*)d_in[8];
  const float* dpb_w1 = (const float*)d_in[9];
  const float* dpb_b1 = (const float*)d_in[10];
  const float* dpb_l1 = (const float*)d_in[11];
  const float* dpb_w2 = (const float*)d_in[12];
  const float* dpb_b2 = (const float*)d_in[13];
  const float* dpb_l2 = (const float*)d_in[14];
  const float* dpb_w3 = (const float*)d_in[15];
  const float* dpb_b3 = (const float*)d_in[16];
  float* out = (float*)d_out;

  // workspace: wqkv_t bf16 [1536x512] | wout_t bf16 [512x512] | bias_mat f32 [64x64]
  bf16_t* wqkv_t  = (bf16_t*)d_ws;
  bf16_t* wout_t  = wqkv_t + 1536 * 512;
  float*  bias_mat = (float*)(wout_t + 512 * 512);

  const int conv_total = 1536 * 512 + 512 * 512;
  convert_weights<<<(conv_total + 255) / 256, 256, 0, stream>>>(w_qkv, w_out, wqkv_t, wout_t);

  dpb_kernel<<<1, 320, 0, stream>>>(dpb_w0, dpb_b0, dpb_l0,
                                    dpb_w1, dpb_b1, dpb_l1,
                                    dpb_w2, dpb_b2, dpb_l2,
                                    dpb_w3, dpb_b3, bias_mat);

  const size_t shmem = 65536 + 196608 + 16384;   // 272 KB dynamic LDS (<= 320 KB WGP)
  (void)hipFuncSetAttribute((const void*)fused_window_attn,
                            hipFuncAttributeMaxDynamicSharedMemorySize, (int)shmem);
  fused_window_attn<<<1024, 256, shmem, stream>>>(x, norm_g, norm_b, wqkv_t, wout_t,
                                                  bias_mat, b_out, out);
}
